// adaptiveFusion_6038724018675
// MI455X (gfx1250) — matrix operand order repaired and checked by reading, not compiled, not run on hardware
//
#include <hip/hip_runtime.h>

#define B_DIM 8
#define T_DIM 128
#define N_DIM 256
#define F_LEN 128
#define LDP   136                 // padded LDS row stride (bf16 elems) -> conflict-free
#define SCALE 0.088388347648318447f   // 1/sqrt(128)
#define EPS_LN 1e-5f
#define NEGM  (-1.0e30f)

typedef __attribute__((ext_vector_type(16))) __bf16 v16bf;
typedef __attribute__((ext_vector_type(8)))  float  v8f;
typedef unsigned short u16;
typedef unsigned int   u32;

__device__ __forceinline__ u16 f2bf(float f) {
  u32 u = __float_as_uint(f);
  u32 r = (u + 0x7FFFu + ((u >> 16) & 1u)) >> 16;   // RNE
  return (u16)r;
}
__device__ __forceinline__ float bf2f(u16 b) {
  return __uint_as_float(((u32)b) << 16);
}

// ---- WMMA fragment loaders (layouts per CDNA5 ISA §7.12.2) ----
// A: 16x32 bf16, lane (r,h): elems 0..7 = K[k0+8h .. +7], elems 8..15 = K[k0+16+8h .. +7]
__device__ __forceinline__ v16bf lda(const u16* S, int m0, int k0, int r, int h) {
  union { uint4 q[2]; v16bf v; } u;
  const u16* row = S + (m0 + r) * LDP + k0 + 8 * h;
  u.q[0] = *(const uint4*)(row);
  u.q[1] = *(const uint4*)(row + 16);
  return u.v;
}
// B: 32x16 bf16 where source holds B^T row-major ([n][k], stride `str`):
// lane (r,h): elems 0..7 = K[k0+8h .. +7], elems 8..15 = K[k0+16+8h .. +7], column n0+r
__device__ __forceinline__ v16bf ldbT(const u16* S, int str, int k0, int n0, int r, int h) {
  union { uint4 q[2]; v16bf v; } u;
  const u16* p = S + (n0 + r) * str + k0 + 8 * h;
  u.q[0] = *(const uint4*)(p);
  u.q[1] = *(const uint4*)(p + 16);
  return u.v;
}

__device__ __forceinline__ void zero8(v8f a[8]) {
#pragma unroll
  for (int j = 0; j < 8; ++j) { v8f z = {0.f,0.f,0.f,0.f,0.f,0.f,0.f,0.f}; a[j] = z; }
}

// One wave computes rows [m0, m0+16) of a 128x128 GEMM: D = A(128xK=128) * B
__device__ __forceinline__ void gemm_strip(v8f acc[8], const u16* A,
                                           const u16* BT, int bstr,
                                           int m0, int r, int h) {
#pragma unroll
  for (int k0 = 0; k0 < 128; k0 += 32) {
    v16bf a = lda(A, m0, k0, r, h);
#pragma unroll
    for (int j = 0; j < 8; ++j) {
      v16bf b = ldbT(BT, bstr, k0, 16 * j, r, h);
      acc[j] = __builtin_amdgcn_wmma_f32_16x16x32_bf16(
          false, a, false, b, (short)0, acc[j], false, false);
    }
  }
}

// reductions within a 16-lane half (rows of a D fragment live across 16 lanes)
__device__ __forceinline__ float redmax16(float x) {
  x = fmaxf(x, __shfl_xor(x, 1)); x = fmaxf(x, __shfl_xor(x, 2));
  x = fmaxf(x, __shfl_xor(x, 4)); x = fmaxf(x, __shfl_xor(x, 8));
  return x;
}
__device__ __forceinline__ float redsum16(float x) {
  x += __shfl_xor(x, 1); x += __shfl_xor(x, 2);
  x += __shfl_xor(x, 4); x += __shfl_xor(x, 8);
  return x;
}

// ---------------- prep: fp32 W[k][n] -> bf16 W^T[n][k] in workspace ----------------
__global__ void prep_weights_kernel(const float* __restrict__ w0, const float* __restrict__ w1,
                                    const float* __restrict__ w2, const float* __restrict__ w3,
                                    const float* __restrict__ w4, const float* __restrict__ w5,
                                    u16* __restrict__ out) {
  const float* wsrc[6] = {w0, w1, w2, w3, w4, w5};
  const float* w = wsrc[blockIdx.x];
  u16* o = out + (size_t)blockIdx.x * (F_LEN * F_LEN);
  for (int idx = threadIdx.x; idx < F_LEN * F_LEN; idx += blockDim.x) {
    int k = idx >> 7, n = idx & 127;
    o[n * F_LEN + k] = f2bf(w[idx]);
  }
}

// ---------------- fused attention block: one workgroup per (b, vertex) ----------------
__global__ void __launch_bounds__(256)
fused_attention_block(const float* __restrict__ xl, const float* __restrict__ xh,
                      const float* __restrict__ te,
                      const float* __restrict__ bq,  const float* __restrict__ bkh,
                      const float* __restrict__ bvh, const float* __restrict__ bo,
                      const float* __restrict__ bf1, const float* __restrict__ bf2,
                      const u16* __restrict__ wT, float* __restrict__ out) {
  extern __shared__ u16 smem[];
  u16* S0 = smem;                 // XL (kept for residual)
  u16* S1 = smem + 1 * 128 * LDP; // XH -> Q -> VAL0
  u16* S2 = smem + 2 * 128 * LDP; // KH -> ATT -> H
  u16* S3 = smem + 3 * 128 * LDP; // VH^T -> VAL1

  const int b   = blockIdx.x >> 8;
  const int n   = blockIdx.x & 255;
  const int tid = threadIdx.x;
  const int w    = tid >> 5;
  const int lane = tid & 31;
  const int r    = lane & 15;
  const int h    = lane >> 4;
  const int m0   = 16 * w;

  const u16* wqT  = wT + 0 * 16384;
  const u16* wkhT = wT + 1 * 16384;
  const u16* wvhT = wT + 2 * 16384;
  const u16* woT  = wT + 3 * 16384;
  const u16* wf1T = wT + 4 * 16384;
  const u16* wf2T = wT + 5 * 16384;

  // ---- Phase A: stream xl,xh (+te) in once, store bf16 in LDS ----
#pragma unroll 4
  for (int it = 0; it < 16; ++it) {
    int t  = it * 8 + (tid >> 5);
    int f4 = (tid & 31) * 4;
    size_t rowg = ((size_t)(b * T_DIM + t) * N_DIM + n) * F_LEN + f4;
    float4 tv = *(const float4*)(te + (size_t)(b * T_DIM + t) * F_LEN + f4);
    float4 a  = *(const float4*)(xl + rowg);
    float4 c  = *(const float4*)(xh + rowg);
    ushort4 pa = { f2bf(a.x + tv.x), f2bf(a.y + tv.y), f2bf(a.z + tv.z), f2bf(a.w + tv.w) };
    ushort4 pc = { f2bf(c.x + tv.x), f2bf(c.y + tv.y), f2bf(c.z + tv.z), f2bf(c.w + tv.w) };
    *(ushort4*)(S0 + t * LDP + f4) = pa;
    *(ushort4*)(S1 + t * LDP + f4) = pc;
  }
  __syncthreads();

  v8f acc[8];

  // ---- P1: KH = relu(XH @ Wkh + bkh) -> S2 row-major ----
  zero8(acc);
  gemm_strip(acc, S1, wkhT, 128, m0, r, h);
#pragma unroll
  for (int j = 0; j < 8; ++j) {
    float bias = bkh[16 * j + r];
#pragma unroll
    for (int v = 0; v < 8; ++v)
      S2[(m0 + v + 8 * h) * LDP + 16 * j + r] = f2bf(fmaxf(acc[j][v] + bias, 0.0f));
  }

  // ---- P2: VH = relu(XH @ Wvh + bvh), stored transposed (VH^T[f][s]) -> S3 ----
  zero8(acc);
  gemm_strip(acc, S1, wvhT, 128, m0, r, h);
#pragma unroll
  for (int j = 0; j < 8; ++j) {
    float bias = bvh[16 * j + r];
    union { u16 s[8]; uint4 q; } u;
#pragma unroll
    for (int v = 0; v < 8; ++v) u.s[v] = f2bf(fmaxf(acc[j][v] + bias, 0.0f));
    *(uint4*)(S3 + (16 * j + r) * LDP + m0 + 8 * h) = u.q;   // contiguous in t
  }
  __syncthreads();  // S2,S3 ready; all XH readers done

  // ---- P3: Q = XL @ Wq + bq -> S1 (overwrites XH) ----
  zero8(acc);
  gemm_strip(acc, S0, wqT, 128, m0, r, h);
#pragma unroll
  for (int j = 0; j < 8; ++j) {
    float bias = bq[16 * j + r];
#pragma unroll
    for (int v = 0; v < 8; ++v)
      S1[(m0 + v + 8 * h) * LDP + 16 * j + r] = f2bf(acc[j][v] + bias);
  }
  __syncthreads();  // Q ready

  // ---- P4: scores = Q @ KH^T ; causal mask + softmax in registers ----
  zero8(acc);
  gemm_strip(acc, S1, S2, LDP, m0, r, h);  // KH row-major acts as B^T source
#pragma unroll
  for (int v = 0; v < 8; ++v) {
    int trow = m0 + v + 8 * h;
    float x[8];
#pragma unroll
    for (int j = 0; j < 8; ++j) {
      int s = 16 * j + r;
      x[j] = (s <= trow) ? acc[j][v] * SCALE : NEGM;
    }
    float mx = x[0];
#pragma unroll
    for (int j = 1; j < 8; ++j) mx = fmaxf(mx, x[j]);
    mx = redmax16(mx);
    float sm = 0.f;
#pragma unroll
    for (int j = 0; j < 8; ++j) { x[j] = __expf(x[j] - mx); sm += x[j]; }
    sm = redsum16(sm);
    float inv = 1.0f / sm;
#pragma unroll
    for (int j = 0; j < 8; ++j) acc[j][v] = x[j] * inv;
  }
  __syncthreads();  // all Q/KH reads done -> S2 reusable
#pragma unroll
  for (int j = 0; j < 8; ++j)
#pragma unroll
    for (int v = 0; v < 8; ++v)
      S2[(m0 + v + 8 * h) * LDP + 16 * j + r] = f2bf(acc[j][v]);  // ATT row-major
  __syncthreads();  // ATT ready

  // ---- P5: VAL0 = ATT @ VH -> S1 (Q dead) ----
  zero8(acc);
  gemm_strip(acc, S2, S3, LDP, m0, r, h);  // B from VH^T rows (contiguous)
#pragma unroll
  for (int j = 0; j < 8; ++j)
#pragma unroll
    for (int v = 0; v < 8; ++v)
      S1[(m0 + v + 8 * h) * LDP + 16 * j + r] = f2bf(acc[j][v]);
  __syncthreads();  // VAL0 ready; VH^T readers done -> S3 reusable

  // ---- P6: VAL1 = LayerNorm(VAL0 @ Wo + bo + XL) -> S3 ----
  zero8(acc);
  gemm_strip(acc, S1, woT, 128, m0, r, h);
  {
    float bias[8];
#pragma unroll
    for (int j = 0; j < 8; ++j) bias[j] = bo[16 * j + r];
#pragma unroll
    for (int v = 0; v < 8; ++v) {
      int trow = m0 + v + 8 * h;
      float x[8], s1 = 0.f, s2 = 0.f;
#pragma unroll
      for (int j = 0; j < 8; ++j) {
        x[j] = acc[j][v] + bias[j] + bf2f(S0[trow * LDP + 16 * j + r]);
        s1 += x[j]; s2 += x[j] * x[j];
      }
      s1 = redsum16(s1); s2 = redsum16(s2);
      float mean = s1 * (1.0f / 128.0f);
      float var  = s2 * (1.0f / 128.0f) - mean * mean;
      float rstd = rsqrtf(var + EPS_LN);
#pragma unroll
      for (int j = 0; j < 8; ++j) acc[j][v] = (x[j] - mean) * rstd;
    }
  }
#pragma unroll
  for (int j = 0; j < 8; ++j)
#pragma unroll
    for (int v = 0; v < 8; ++v)
      S3[(m0 + v + 8 * h) * LDP + 16 * j + r] = f2bf(acc[j][v]);
  __syncthreads();  // VAL1 ready

  // ---- P7: H = relu(VAL1 @ Wf1 + bf1) -> S2 (ATT dead) ----
  zero8(acc);
  gemm_strip(acc, S3, wf1T, 128, m0, r, h);
#pragma unroll
  for (int j = 0; j < 8; ++j) {
    float bias = bf1[16 * j + r];
#pragma unroll
    for (int v = 0; v < 8; ++v)
      S2[(m0 + v + 8 * h) * LDP + 16 * j + r] = f2bf(fmaxf(acc[j][v] + bias, 0.0f));
  }
  __syncthreads();  // H ready

  // ---- P8: OUT = LayerNorm(H @ Wf2 + bf2 + VAL1) -> global fp32 ----
  zero8(acc);
  gemm_strip(acc, S2, wf2T, 128, m0, r, h);
  {
    float bias[8];
#pragma unroll
    for (int j = 0; j < 8; ++j) bias[j] = bf2[16 * j + r];
#pragma unroll
    for (int v = 0; v < 8; ++v) {
      int trow = m0 + v + 8 * h;
      float x[8], s1 = 0.f, s2 = 0.f;
#pragma unroll
      for (int j = 0; j < 8; ++j) {
        x[j] = acc[j][v] + bias[j] + bf2f(S3[trow * LDP + 16 * j + r]);
        s1 += x[j]; s2 += x[j] * x[j];
      }
      s1 = redsum16(s1); s2 = redsum16(s2);
      float mean = s1 * (1.0f / 128.0f);
      float var  = s2 * (1.0f / 128.0f) - mean * mean;
      float rstd = rsqrtf(var + EPS_LN);
      size_t rowg = ((size_t)(b * T_DIM + trow) * N_DIM + n) * F_LEN;
#pragma unroll
      for (int j = 0; j < 8; ++j)
        out[rowg + 16 * j + r] = (x[j] - mean) * rstd;
    }
  }
}

extern "C" void kernel_launch(void* const* d_in, const int* in_sizes, int n_in,
                              void* d_out, int out_size, void* d_ws, size_t ws_size,
                              hipStream_t stream) {
  (void)in_sizes; (void)n_in; (void)out_size; (void)ws_size;
  const float* xl  = (const float*)d_in[0];
  const float* xh  = (const float*)d_in[1];
  const float* te  = (const float*)d_in[2];
  const float* Wq  = (const float*)d_in[3];   const float* bq  = (const float*)d_in[4];
  const float* Wkh = (const float*)d_in[5];   const float* bkh = (const float*)d_in[6];
  const float* Wvh = (const float*)d_in[7];   const float* bvh = (const float*)d_in[8];
  const float* Wo  = (const float*)d_in[9];   const float* bo  = (const float*)d_in[10];
  const float* Wf1 = (const float*)d_in[11];  const float* bf1 = (const float*)d_in[12];
  const float* Wf2 = (const float*)d_in[13];  const float* bf2 = (const float*)d_in[14];
  u16* wT = (u16*)d_ws;  // 6 * 128*128 bf16 = 192 KB

  prep_weights_kernel<<<6, 256, 0, stream>>>(Wq, Wkh, Wvh, Wo, Wf1, Wf2, wT);

  size_t shmem = (size_t)4 * 128 * LDP * sizeof(u16);  // 139264 B of the 320 KB WGP LDS
  (void)hipFuncSetAttribute((const void*)fused_attention_block,
                            hipFuncAttributeMaxDynamicSharedMemorySize, (int)shmem);
  fused_attention_block<<<dim3(B_DIM * N_DIM), dim3(256), shmem, stream>>>(
      xl, xh, te, bq, bkh, bvh, bo, bf1, bf2, wT, (float*)d_out);
}
